// MEGNetBlock_66005057405562
// MI455X (gfx1250) — matrix operand order repaired and checked by reading, not compiled, not run on hardware
//
#include <hip/hip_runtime.h>
#include <hip/hip_bf16.h>
#include <stdint.h>

// ---------------------------------------------------------------------------
// MEGNet block, fused gather+GEMM via v_wmma_f32_16x16x32_f16 (gfx1250 wave32)
// M-blocked: each wave computes 2 x (16x128) output tiles sharing B fragments,
// halving packed-weight (L2) traffic per WMMA.
// ---------------------------------------------------------------------------

typedef __attribute__((ext_vector_type(16))) _Float16 v16h;
typedef __attribute__((ext_vector_type(8)))  _Float16 v8h;
typedef __attribute__((ext_vector_type(4)))  _Float16 v4h;
typedef __attribute__((ext_vector_type(8)))  float    v8f;

#define HDIM 128

__device__ __forceinline__ float softplusf(float v) {
    return v > 20.f ? v : log1pf(expf(v));
}

__device__ __forceinline__ v4h cvt4(float a, float b, float c, float d) {
    v4h h; h[0] = (_Float16)a; h[1] = (_Float16)b; h[2] = (_Float16)c; h[3] = (_Float16)d;
    return h;
}

// A fragment loader (ISA 7.12.2, 16-bit A 16x32): lane L holds row M=L&15;
// lanes<16 hold K=0..7 & 16..23, lanes>=16 hold K=8..15 & 24..31 per k-block.
__device__ __forceinline__ v16h a_frag(const _Float16* __restrict__ rowA, int off) {
    v8h lo = *(const v8h*)(rowA + off);
    v8h hi = *(const v8h*)(rowA + off + 16);
    return __builtin_shufflevector(lo, hi, 0,1,2,3,4,5,6,7,8,9,10,11,12,13,14,15);
}

// Single-M-tile GEMM (used by the tiny global-update kernel).
// B packed layout: v16h index = kb*256 + t*32 + lane; lane holds column
// n = t*16 + (lane&15), K = kb*32 + (lane>=16?8:0) + {0..7, 16..23}.
template <int KB>
__device__ __forceinline__ void wave_gemm1(const _Float16* __restrict__ rowA,
                                           const v16h* __restrict__ Bp,
                                           int lane, v8f acc[8]) {
    const int half = (lane >> 4) << 3;
    for (int kb = 0; kb < KB; ++kb) {
        v16h a = a_frag(rowA, kb * 32 + half);
        const v16h* bb = Bp + (size_t)kb * 256 + lane;
#pragma unroll
        for (int t = 0; t < 8; ++t) {
            acc[t] = __builtin_amdgcn_wmma_f32_16x16x32_f16(
                false, a, false, bb[t * 32], (short)0, acc[t], false, false);
        }
    }
}

// Dual-M-tile GEMM: two A rows share each B fragment (2x B reuse).
template <int KB>
__device__ __forceinline__ void wave_gemm2(const _Float16* __restrict__ rowA0,
                                           const _Float16* __restrict__ rowA1,
                                           const v16h* __restrict__ Bp,
                                           int lane, v8f acc0[8], v8f acc1[8]) {
    const int half = (lane >> 4) << 3;
    for (int kb = 0; kb < KB; ++kb) {
        v16h a0 = a_frag(rowA0, kb * 32 + half);
        v16h a1 = a_frag(rowA1, kb * 32 + half);
        const v16h* bb = Bp + (size_t)kb * 256 + lane;
#pragma unroll
        for (int t = 0; t < 8; ++t) {
            v16h b = bb[t * 32];
            acc0[t] = __builtin_amdgcn_wmma_f32_16x16x32_f16(
                false, a0, false, b, (short)0, acc0[t], false, false);
            acc1[t] = __builtin_amdgcn_wmma_f32_16x16x32_f16(
                false, a1, false, b, (short)0, acc1[t], false, false);
        }
    }
}

// ---------------------------------------------------------------------------
// Weight packing: W [K,128] f32 row-major -> f16 B fragments as above.
// flat idx bits: [3:0]=slot s, [8:4]=lane, [11:9]=tile t, [..:12]=kb
// ---------------------------------------------------------------------------
__global__ void pack_w_kernel(const float* __restrict__ W, _Float16* __restrict__ out,
                              int total) {
    int idx = blockIdx.x * blockDim.x + threadIdx.x;
    if (idx >= total) return;
    int s    = idx & 15;
    int lane = (idx >> 4) & 31;
    int t    = (idx >> 9) & 7;
    int kb   = idx >> 12;
    int k = kb * 32 + ((lane >> 4) << 3) + (s & 7) + ((s >> 3) << 4);
    int n = t * 16 + (lane & 15);
    out[idx] = (_Float16)W[(size_t)k * HDIM + n];
}

// ---------------------------------------------------------------------------
// Segment counts (f32 so we can reuse atomicAdd path and divide directly)
// ---------------------------------------------------------------------------
__global__ void count_kernel(const long long* __restrict__ eidx,
                             const long long* __restrict__ batch,
                             float* __restrict__ e_cnt, float* __restrict__ ge_cnt,
                             float* __restrict__ gv_cnt, int E, int N) {
    int i = blockIdx.x * blockDim.x + threadIdx.x;
    if (i < E) {
        long long rn = eidx[i];
        atomicAdd(e_cnt + rn, 1.f);
        atomicAdd(ge_cnt + batch[rn], 1.f);
    }
    if (i < N) {
        atomicAdd(gv_cnt + batch[i], 1.f);
    }
}

// ---------------------------------------------------------------------------
// Edge update: 256 edges/block (8 waves x 32 rows). Fused gather-concat
// [x[row] | x[col] | edge_attr | u[batch[row]]] -> f16 LDS rows -> 2x GEMM.
// ---------------------------------------------------------------------------
#define SE 520  // f16 row stride (512 + 8 pad); 1040B, 16B aligned

__global__ __launch_bounds__(256) void edge_kernel(
    const float* __restrict__ x, const long long* __restrict__ eidx,
    const float* __restrict__ ea, const float* __restrict__ u,
    const long long* __restrict__ batch,
    const float* __restrict__ be1, const float* __restrict__ be2,
    const _Float16* __restrict__ Wp1, const _Float16* __restrict__ Wp2,
    float* __restrict__ outE, float* __restrict__ e_sum,
    float* __restrict__ ge_sum, int E) {
    extern __shared__ _Float16 smem[];
    __shared__ int s_rn[256], s_gr[256];

    const int lane  = threadIdx.x & 31;
    const int wave  = threadIdx.x >> 5;
    const int r     = lane & 15;
    const int hh    = lane >> 4;            // 0/1: which 64-col half this lane stages
    const int wbase = wave * 32;            // first row of this wave's 32-row block

    // ---- stage 2x16 rows, 2 lanes per row ----
#pragma unroll
    for (int rs = 0; rs < 2; ++rs) {
        const int rowi = wbase + rs * 16 + r;
        const long long e = (long long)blockIdx.x * 256 + rowi;
        const bool valid = e < E;
        long long rn = 0, cn = 0, g = 0;
        if (valid) { rn = eidx[e]; cn = eidx[(long long)E + e]; g = batch[rn]; }
        if (hh == 0) { s_rn[rowi] = (int)rn; s_gr[rowi] = (int)g; }

        _Float16* row = smem + (size_t)rowi * SE;
        const float* srcs[4] = { x + rn * HDIM, x + cn * HDIM, ea + e * HDIM, u + g * HDIM };
#pragma unroll
        for (int s = 0; s < 4; ++s) {
            const float* src = srcs[s] + hh * 64;
            _Float16* dst = row + s * 128 + hh * 64;
            for (int i = 0; i < 16; ++i) {
                float4 f = valid ? ((const float4*)src)[i] : make_float4(0.f, 0.f, 0.f, 0.f);
                ((v4h*)dst)[i] = cvt4(f.x, f.y, f.z, f.w);
            }
        }
    }
    __syncthreads();

    const _Float16* rowA0 = smem + (size_t)(wbase + r) * SE;
    const _Float16* rowA1 = smem + (size_t)(wbase + 16 + r) * SE;

    // GEMM1: [32x512] x [512x128], K=512 -> 16 k-blocks
    v8f acc0[8] = {}, acc1[8] = {};
    wave_gemm2<16>(rowA0, rowA1, (const v16h*)Wp1, lane, acc0, acc1);

    // epilogue1: bias + softplus -> f16 h1 into cols [0,128) of own tile rows
    const int mo = hh * 8;  // D layout: lanes>=16 hold M=q+8
#pragma unroll
    for (int t = 0; t < 8; ++t) {
        const int n = t * 16 + r;
        const float b1 = be1[n];
#pragma unroll
        for (int q = 0; q < 8; ++q) {
            smem[(size_t)(wbase + mo + q) * SE + n]      = (_Float16)softplusf(acc0[t][q] + b1);
            smem[(size_t)(wbase + 16 + mo + q) * SE + n] = (_Float16)softplusf(acc1[t][q] + b1);
        }
    }
    asm volatile("s_wait_dscnt 0" ::: "memory");

    // GEMM2: [32x128] x [128x128], K=128 -> 4 k-blocks
    v8f acc2a[8] = {}, acc2b[8] = {};
    wave_gemm2<4>(rowA0, rowA1, (const v16h*)Wp2, lane, acc2a, acc2b);

#pragma unroll
    for (int t = 0; t < 8; ++t) {
        const int n = t * 16 + r;
        const float b2 = be2[n];
#pragma unroll
        for (int rs = 0; rs < 2; ++rs) {
            v8f* ac = rs ? acc2b : acc2a;
#pragma unroll
            for (int q = 0; q < 8; ++q) {
                const int rowi = wbase + rs * 16 + mo + q;
                const long long e2 = (long long)blockIdx.x * 256 + rowi;
                if (e2 < E) {
                    float v = softplusf(ac[t][q] + b2);
                    outE[e2 * HDIM + n] = v;
                    atomicAdd(e_sum + (size_t)s_rn[rowi] * HDIM + n, v);
                    atomicAdd(ge_sum + (size_t)s_gr[rowi] * HDIM + n, v);
                }
            }
        }
    }
}

// ---------------------------------------------------------------------------
// Node update: 256 nodes/block (8 waves x 32 rows). [x | e_aggr | u[batch]]
// (e_aggr = e_sum/cnt folded into staging) -> GEMM K=384 -> GEMM K=128.
// ---------------------------------------------------------------------------
#define SV 400  // f16 row stride (384 + 16 pad); 800B, 16B aligned

__global__ __launch_bounds__(256) void node_kernel(
    const float* __restrict__ x, const long long* __restrict__ batch,
    const float* __restrict__ u, const float* __restrict__ e_sum,
    const float* __restrict__ e_cnt,
    const float* __restrict__ bv1, const float* __restrict__ bv2,
    const _Float16* __restrict__ Wp1, const _Float16* __restrict__ Wp2,
    float* __restrict__ outX, float* __restrict__ gv_sum, int N) {
    extern __shared__ _Float16 smem[];
    __shared__ int s_g[256];

    const int lane  = threadIdx.x & 31;
    const int wave  = threadIdx.x >> 5;
    const int r     = lane & 15;
    const int hh    = lane >> 4;
    const int wbase = wave * 32;

#pragma unroll
    for (int rs = 0; rs < 2; ++rs) {
        const int rowi = wbase + rs * 16 + r;
        const long long i = (long long)blockIdx.x * 256 + rowi;
        const bool valid = i < N;
        long long g = 0;
        float inv = 1.f;
        if (valid) { g = batch[i]; inv = 1.f / fmaxf(e_cnt[i], 1.f); }
        if (hh == 0) s_g[rowi] = (int)g;

        _Float16* row = smem + (size_t)rowi * SV;
        const float* srcs[3]  = { x + i * HDIM, e_sum + i * HDIM, u + g * HDIM };
        const float  scale[3] = { 1.f, inv, 1.f };
#pragma unroll
        for (int s = 0; s < 3; ++s) {
            const float* src = srcs[s] + hh * 64;
            const float sc = scale[s];
            _Float16* dst = row + s * 128 + hh * 64;
            for (int j = 0; j < 16; ++j) {
                float4 f = valid ? ((const float4*)src)[j] : make_float4(0.f, 0.f, 0.f, 0.f);
                ((v4h*)dst)[j] = cvt4(f.x * sc, f.y * sc, f.z * sc, f.w * sc);
            }
        }
    }
    __syncthreads();

    const _Float16* rowA0 = smem + (size_t)(wbase + r) * SV;
    const _Float16* rowA1 = smem + (size_t)(wbase + 16 + r) * SV;

    v8f acc0[8] = {}, acc1[8] = {};
    wave_gemm2<12>(rowA0, rowA1, (const v16h*)Wp1, lane, acc0, acc1);  // K = 384

    const int mo = hh * 8;
#pragma unroll
    for (int t = 0; t < 8; ++t) {
        const int n = t * 16 + r;
        const float b1 = bv1[n];
#pragma unroll
        for (int q = 0; q < 8; ++q) {
            smem[(size_t)(wbase + mo + q) * SV + n]      = (_Float16)softplusf(acc0[t][q] + b1);
            smem[(size_t)(wbase + 16 + mo + q) * SV + n] = (_Float16)softplusf(acc1[t][q] + b1);
        }
    }
    asm volatile("s_wait_dscnt 0" ::: "memory");

    v8f acc2a[8] = {}, acc2b[8] = {};
    wave_gemm2<4>(rowA0, rowA1, (const v16h*)Wp2, lane, acc2a, acc2b);  // K = 128

#pragma unroll
    for (int t = 0; t < 8; ++t) {
        const int n = t * 16 + r;
        const float b2 = bv2[n];
#pragma unroll
        for (int rs = 0; rs < 2; ++rs) {
            v8f* ac = rs ? acc2b : acc2a;
#pragma unroll
            for (int q = 0; q < 8; ++q) {
                const int rowi = wbase + rs * 16 + mo + q;
                const long long i2 = (long long)blockIdx.x * 256 + rowi;
                if (i2 < N) {
                    float v = softplusf(ac[t][q] + b2);
                    outX[i2 * HDIM + n] = v;
                    atomicAdd(gv_sum + (size_t)s_g[rowi] * HDIM + n, v);
                }
            }
        }
    }
}

// ---------------------------------------------------------------------------
// Global update: one block of 4 waves covers G=64 graphs (single-tile path).
// ---------------------------------------------------------------------------
__global__ __launch_bounds__(128) void global_kernel(
    const float* __restrict__ u,
    const float* __restrict__ ge_sum, const float* __restrict__ ge_cnt,
    const float* __restrict__ gv_sum, const float* __restrict__ gv_cnt,
    const float* __restrict__ bu1, const float* __restrict__ bu2,
    const _Float16* __restrict__ Wp1, const _Float16* __restrict__ Wp2,
    float* __restrict__ outU, int G) {
    extern __shared__ _Float16 smem[];

    const int lane  = threadIdx.x & 31;
    const int wave  = threadIdx.x >> 5;
    const int r     = lane & 15;
    const int hh    = lane >> 4;
    const int tbase = wave * 16;

    const long long gi = (long long)blockIdx.x * 64 + tbase + r;
    const bool valid = gi < G;
    float inv_e = 1.f, inv_v = 1.f;
    if (valid) {
        inv_e = 1.f / fmaxf(ge_cnt[gi], 1.f);
        inv_v = 1.f / fmaxf(gv_cnt[gi], 1.f);
    }

    _Float16* row = smem + (size_t)(tbase + r) * SV;
    const float* srcs[3]  = { u + gi * HDIM, ge_sum + gi * HDIM, gv_sum + gi * HDIM };
    const float  scale[3] = { 1.f, inv_e, inv_v };
#pragma unroll
    for (int s = 0; s < 3; ++s) {
        const float* src = srcs[s] + hh * 64;
        const float sc = scale[s];
        _Float16* dst = row + s * 128 + hh * 64;
        for (int j = 0; j < 16; ++j) {
            float4 f = valid ? ((const float4*)src)[j] : make_float4(0.f, 0.f, 0.f, 0.f);
            ((v4h*)dst)[j] = cvt4(f.x * sc, f.y * sc, f.z * sc, f.w * sc);
        }
    }
    __syncthreads();

    v8f acc[8] = {};
    wave_gemm1<12>(row, (const v16h*)Wp1, lane, acc);

    const int mo = hh * 8;
#pragma unroll
    for (int t = 0; t < 8; ++t) {
        const int n = t * 16 + r;
        const float b1 = bu1[n];
#pragma unroll
        for (int q = 0; q < 8; ++q) {
            smem[(size_t)(tbase + mo + q) * SV + n] = (_Float16)softplusf(acc[t][q] + b1);
        }
    }
    asm volatile("s_wait_dscnt 0" ::: "memory");

    v8f acc2[8] = {};
    wave_gemm1<4>(row, (const v16h*)Wp2, lane, acc2);

#pragma unroll
    for (int t = 0; t < 8; ++t) {
        const int n = t * 16 + r;
        const float b2 = bu2[n];
#pragma unroll
        for (int q = 0; q < 8; ++q) {
            const long long g2 = (long long)blockIdx.x * 64 + tbase + mo + q;
            if (g2 < G) {
                outU[g2 * HDIM + n] = softplusf(acc2[t][q] + b2);
            }
        }
    }
}

// ---------------------------------------------------------------------------
extern "C" void kernel_launch(void* const* d_in, const int* in_sizes, int n_in,
                              void* d_out, int out_size, void* d_ws, size_t ws_size,
                              hipStream_t stream) {
    const float*     x     = (const float*)d_in[0];
    const long long* eidx  = (const long long*)d_in[1];
    const float*     ea    = (const float*)d_in[2];
    const float*     u     = (const float*)d_in[3];
    const long long* batch = (const long long*)d_in[4];
    const float* We1 = (const float*)d_in[5];
    const float* be1 = (const float*)d_in[6];
    const float* We2 = (const float*)d_in[7];
    const float* be2 = (const float*)d_in[8];
    const float* Wv1 = (const float*)d_in[9];
    const float* bv1 = (const float*)d_in[10];
    const float* Wv2 = (const float*)d_in[11];
    const float* bv2 = (const float*)d_in[12];
    const float* Wu1 = (const float*)d_in[13];
    const float* bu1 = (const float*)d_in[14];
    const float* Wu2 = (const float*)d_in[15];
    const float* bu2 = (const float*)d_in[16];

    const int N = in_sizes[0] / HDIM;
    const int E = in_sizes[2] / HDIM;
    const int G = in_sizes[3] / HDIM;

    // ---- workspace carve (256B aligned regions) ----
    char* p = (char*)d_ws;
    auto take = [&](size_t bytes) {
        void* q = (void*)p;
        p += (bytes + 255) & ~(size_t)255;
        return q;
    };
    _Float16* WpE1 = (_Float16*)take((size_t)512 * HDIM * 2);
    _Float16* WpE2 = (_Float16*)take((size_t)128 * HDIM * 2);
    _Float16* WpV1 = (_Float16*)take((size_t)384 * HDIM * 2);
    _Float16* WpV2 = (_Float16*)take((size_t)128 * HDIM * 2);
    _Float16* WpU1 = (_Float16*)take((size_t)384 * HDIM * 2);
    _Float16* WpU2 = (_Float16*)take((size_t)128 * HDIM * 2);
    float* e_sum  = (float*)take((size_t)N * HDIM * 4);
    float* e_cnt  = (float*)take((size_t)N * 4);
    float* ge_sum = (float*)take((size_t)G * HDIM * 4);
    float* ge_cnt = (float*)take((size_t)G * 4);
    float* gv_sum = (float*)take((size_t)G * HDIM * 4);
    float* gv_cnt = (float*)take((size_t)G * 4);

    hipMemsetAsync(e_sum,  0, (size_t)N * HDIM * 4, stream);
    hipMemsetAsync(e_cnt,  0, (size_t)N * 4, stream);
    hipMemsetAsync(ge_sum, 0, (size_t)G * HDIM * 4, stream);
    hipMemsetAsync(ge_cnt, 0, (size_t)G * 4, stream);
    hipMemsetAsync(gv_sum, 0, (size_t)G * HDIM * 4, stream);
    hipMemsetAsync(gv_cnt, 0, (size_t)G * 4, stream);

    // ---- pack weights to f16 WMMA B fragments ----
    auto pack = [&](const float* W, _Float16* out, int K) {
        int total = K * HDIM;
        pack_w_kernel<<<(total + 255) / 256, 256, 0, stream>>>(W, out, total);
    };
    pack(We1, WpE1, 512);
    pack(We2, WpE2, 128);
    pack(Wv1, WpV1, 384);
    pack(Wv2, WpV2, 128);
    pack(Wu1, WpU1, 384);
    pack(Wu2, WpU2, 128);

    // ---- segment counts ----
    int M = E > N ? E : N;
    count_kernel<<<(M + 255) / 256, 256, 0, stream>>>(eidx, batch, e_cnt, ge_cnt,
                                                      gv_cnt, E, N);

    float* outX = (float*)d_out;
    float* outE = outX + (size_t)N * HDIM;
    float* outU = outE + (size_t)E * HDIM;

    // ---- edge update (also accumulates e_sum / ge_sum) ----
    edge_kernel<<<(E + 255) / 256, 256, (size_t)256 * SE * 2, stream>>>(
        x, eidx, ea, u, batch, be1, be2, WpE1, WpE2, outE, e_sum, ge_sum, E);

    // ---- node update (also accumulates gv_sum) ----
    node_kernel<<<(N + 255) / 256, 256, (size_t)256 * SV * 2, stream>>>(
        x, batch, u, e_sum, e_cnt, bv1, bv2, WpV1, WpV2, outX, gv_sum, N);

    // ---- global update ----
    global_kernel<<<(G + 63) / 64, 128, (size_t)64 * SV * 2, stream>>>(
        u, ge_sum, ge_cnt, gv_sum, gv_cnt, bu1, bu2, WpU1, WpU2, outU, G);
}
